// MultiheadVAA_79929341378888
// MI455X (gfx1250) — matrix operand order repaired and checked by reading, not compiled, not run on hardware
//
#include <hip/hip_runtime.h>
#include <stdint.h>
#include <stddef.h>

// ---------------------------------------------------------------------------
// Problem constants (match the reference)
// ---------------------------------------------------------------------------
#define B_     4
#define C_     256
#define H_     96
#define W_     96
#define HW_    (H_ * W_)          // 9216
#define HEADS_ 8
#define CH_    (C_ / HEADS_)      // 32

typedef __attribute__((ext_vector_type(16))) __bf16 v16bf;
typedef __attribute__((ext_vector_type(8)))  float  v8f;

// 8 dwords == 16 packed bf16 == one WMMA 16x16x32 A or B fragment (wave32).
union Frag16 {
    v16bf    v;
    uint32_t u[8];
};

// float -> bf16 with round-to-nearest-even.
__device__ __forceinline__ uint16_t f2bf(float f) {
    union { float f; uint32_t u; } x; x.f = f;
    uint32_t r = x.u + 0x7FFFu + ((x.u >> 16) & 1u);
    return (uint16_t)(r >> 16);
}

// ---------------------------------------------------------------------------
// Kernel 1a: convert wq / wo (fp32, 256x256 row-major [o][c]) to bf16.
// ---------------------------------------------------------------------------
__global__ __launch_bounds__(256) void cvt_weights(const float* __restrict__ wq,
                                                   const float* __restrict__ wo,
                                                   uint16_t* __restrict__ wqb,
                                                   uint16_t* __restrict__ wob) {
    int i = blockIdx.x * 256 + threadIdx.x;
    if (i < C_ * C_) {
        wqb[i] = f2bf(wq[i]);
        wob[i] = f2bf(wo[i]);
    }
}

// ---------------------------------------------------------------------------
// Kernel 1b: transpose-convert q/k/v: fp32 [src][b][c][hw] -> bf16 [src][b][hw][c]
// LDS-tiled 64x64 transpose (pad 2 u16 -> 33-dword row stride, conflict-free).
// With K contiguous per pixel, every GEMM B-fragment becomes two b128 loads,
// exactly like the A operand -- no per-element packing in the hot loop.
// ---------------------------------------------------------------------------
__global__ __launch_bounds__(256) void cvt_acts_t(const float* __restrict__ q,
                                                  const float* __restrict__ k,
                                                  const float* __restrict__ v,
                                                  uint16_t* __restrict__ xbT) {
    __shared__ uint16_t tile[64][66];   // [c_local][hw_local], padded

    const int z   = blockIdx.z;         // src * B_ + b
    const int src = z / B_;
    const int b   = z % B_;
    const float* X = (src == 0 ? q : (src == 1 ? k : v)) + (size_t)b * C_ * HW_;
    uint16_t* dstZ = xbT + (size_t)z * C_ * HW_;

    const int hw0 = blockIdx.x * 64;
    const int c0  = blockIdx.y * 64;
    const int tid = threadIdx.x;

    // Read 64x64 fp32 tile, coalesced along hw; convert; stage in LDS.
#pragma unroll
    for (int i = 0; i < 16; ++i) {
        const int lin = i * 256 + tid;
        const int cl  = lin >> 6;           // 0..63
        const int hwl = lin & 63;           // 0..63
        tile[cl][hwl] = f2bf(X[(size_t)(c0 + cl) * HW_ + hw0 + hwl]);
    }
    __syncthreads();

    // Write transposed, coalesced along c (contiguous bf16).
#pragma unroll
    for (int i = 0; i < 16; ++i) {
        const int lin = i * 256 + tid;
        const int hwl = lin >> 6;
        const int cl  = lin & 63;
        dstZ[(size_t)(hw0 + hwl) * C_ + c0 + cl] = tile[cl][hwl];
    }
}

// ---------------------------------------------------------------------------
// WMMA fragment layouts (wave32, 16x16x32 bf16), per ISA 7.12.2:
//
// A (16x32, M x K): lane: m = lane & 15, half = lane >> 4
//   vgpr v, sub s: k = ((v>>2)<<4) + half*8 + ((v&3)<<1) + s
//   -> per lane two contiguous 16B runs (2x global_load_b128).
//
// B (32x16, K x N): lane: n = lane & 15, half = lane >> 4
//   vgpr v, sub s: k = ((v>>2)<<4) + half*8 + ((v&3)<<1) + s  (same as A)
//   -> in the TRANSPOSED activation layout [n][k] this is two contiguous
//      16B K-runs per lane, zero packing ALU.
//
// C/D (16x16 f32): lane: n = lane & 15; vgpr r: m = r + 8*(lane>>4)
// ---------------------------------------------------------------------------

__device__ __forceinline__ void load_a_frag(Frag16& a, const uint16_t* __restrict__ wrow,
                                            int kt, int half) {
#pragma unroll
    for (int r = 0; r < 8; ++r) {
        int kk = ((r >> 2) << 4) + (half << 3) + ((r & 3) << 1);
        a.u[r] = *(const uint32_t*)(wrow + kt + kk);   // 4B-aligned: kk even
    }
}

__device__ __forceinline__ void load_b_frag_t(Frag16& bf, const uint16_t* __restrict__ xrow,
                                              int kt, int half) {
#pragma unroll
    for (int r = 0; r < 8; ++r) {
        bf.u[r] = *(const uint32_t*)(xrow + kt + ((r >> 2) << 4) + (half << 3) + ((r & 3) << 1));
    }
}

// ---------------------------------------------------------------------------
// Generic bf16 WMMA GEMM body: D[m, n] (+= bias) = sum_k W[m,k] * XT[n,k]
// Wave tile: 64 M x 32 N  (4 A-frags x 2 B-frags -> 8 WMMAs / K-step).
// Workgroup: 8 waves tiled 2(M) x 4(N) -> 128 x 128 tile.
// ---------------------------------------------------------------------------
__device__ __forceinline__ void gemm_tile(const uint16_t* __restrict__ Wb,  // [M][K] bf16
                                          const uint16_t* __restrict__ XT,  // [HW][K] bf16
                                          const float* __restrict__ bias,   // nullptr or [M]
                                          float* __restrict__ D,            // [M][HW] f32
                                          int mBlock, int nBlock) {
    const int lane = threadIdx.x & 31;
    const int wave = threadIdx.x >> 5;
    const int half = lane >> 4;
    const int lm   = lane & 15;

    const int mBase = mBlock * 128 + (wave & 1) * 64;
    const int nBase = nBlock * 128 + (wave >> 1) * 32;

    const uint16_t* wrow[4];
#pragma unroll
    for (int i = 0; i < 4; ++i)
        wrow[i] = Wb + (size_t)(mBase + i * 16 + lm) * C_;

    const uint16_t* xrow0 = XT + (size_t)(nBase + lm) * C_;
    const uint16_t* xrow1 = XT + (size_t)(nBase + 16 + lm) * C_;

    v8f acc[4][2];
#pragma unroll
    for (int i = 0; i < 4; ++i)
#pragma unroll
        for (int nb = 0; nb < 2; ++nb) acc[i][nb] = v8f{};

    for (int kt = 0; kt < C_; kt += 32) {
        if (kt + 32 < C_)   // prefetch next K-run of activations
            __builtin_prefetch(xrow0 + kt + 32 + (half << 4), 0, 0);

        Frag16 b0, b1;
        load_b_frag_t(b0, xrow0, kt, half);
        load_b_frag_t(b1, xrow1, kt, half);

#pragma unroll
        for (int i = 0; i < 4; ++i) {
            Frag16 a;
            load_a_frag(a, wrow[i], kt, half);
            acc[i][0] = __builtin_amdgcn_wmma_f32_16x16x32_bf16(
                false, a.v, false, b0.v, (short)0, acc[i][0], false, false);
            acc[i][1] = __builtin_amdgcn_wmma_f32_16x16x32_bf16(
                false, a.v, false, b1.v, (short)0, acc[i][1], false, false);
        }
    }

#pragma unroll
    for (int i = 0; i < 4; ++i) {
#pragma unroll
        for (int nb = 0; nb < 2; ++nb) {
            const int n = nBase + nb * 16 + lm;
#pragma unroll
            for (int r = 0; r < 8; ++r) {
                const int m = mBase + i * 16 + r + (half << 3);
                float val = acc[i][nb][r];
                if (bias) val += bias[m];
                D[(size_t)m * HW_ + n] = val;
            }
        }
    }
}

// ---------------------------------------------------------------------------
// Kernel 2: projection GEMM over transposed bf16 activations.
// grid: x = N blocks (HW/128 = 72), y = M blocks (C/128 = 2), z = src*B+b (12)
// ---------------------------------------------------------------------------
__global__ __launch_bounds__(256) void proj_gemm(const uint16_t* __restrict__ xbT,
                                                 const uint16_t* __restrict__ wqb,
                                                 float* __restrict__ proj) {
    const int z = blockIdx.z;   // src * B_ + b
    gemm_tile(wqb,
              xbT + (size_t)z * C_ * HW_,
              nullptr,
              proj + (size_t)z * C_ * HW_,
              blockIdx.y, blockIdx.x);
}

// ---------------------------------------------------------------------------
// Kernel 3: per-head 3x3 local attention (RATE=1 -> kpe=1, PE = scale + 9x9).
// One thread per (head-batch n, pixel p). Lanes map to consecutive w ->
// channel-strided loads coalesce across the wave.
// proj layout: [src][b][c][hw];  output obT: bf16 [b][hw][C] (transposed!)
// ---------------------------------------------------------------------------
__global__ __launch_bounds__(256) void attn(const float* __restrict__ proj,
                                            const float* __restrict__ pe_dw,
                                            const float* __restrict__ pe_pw,
                                            uint16_t* __restrict__ obT) {
    __shared__ float s_dw[9];
    __shared__ float s_pw[81];
    const int tid = threadIdx.x;
    if (tid < 9)  s_dw[tid] = pe_dw[tid];
    if (tid < 81) s_pw[tid] = pe_pw[tid];
    __syncthreads();

    const int gid = blockIdx.x * 256 + tid;           // 1152 * 256 == 294912 exactly
    const int n = gid / HW_;                          // head-batch index (b*8 + head)
    const int p = gid - n * HW_;
    const int h = p / W_;
    const int w = p - h * W_;
    const int b    = n >> 3;
    const int head = n & 7;

    const float* qh = proj + ((size_t)(0 * B_ + b) * C_ + head * CH_) * HW_;
    const float* kh = proj + ((size_t)(1 * B_ + b) * C_ + head * CH_) * HW_;
    const float* vh = proj + ((size_t)(2 * B_ + b) * C_ + head * CH_) * HW_;

    float qv[CH_];
#pragma unroll
    for (int c = 0; c < CH_; ++c) qv[c] = qh[(size_t)c * HW_ + p];

    // 3x3 neighborhood scores (zero padding -> score 0, matching unfold).
    float f[9];
#pragma unroll
    for (int dk = 0; dk < 9; ++dk) {
        const int hh = h + dk / 3 - 1;
        const int ww = w + dk % 3 - 1;
        float acc = 0.f;
        if (hh >= 0 && hh < H_ && ww >= 0 && ww < W_) {
            const int pk = hh * W_ + ww;
#pragma unroll
            for (int c = 0; c < CH_; ++c) acc += qv[c] * kh[(size_t)c * HW_ + pk];
        }
        f[dk] = acc;
    }

    // PE: depthwise 1x1 scale then 9x9 mixing.
    float g[9];
#pragma unroll
    for (int j = 0; j < 9; ++j) {
        float a = 0.f;
#pragma unroll
        for (int i = 0; i < 9; ++i) a += s_pw[j * 9 + i] * (s_dw[i] * f[i]);
        g[j] = a;
    }

    // Softmax over 9 positions.
    float mx = g[0];
#pragma unroll
    for (int j = 1; j < 9; ++j) mx = fmaxf(mx, g[j]);
    float e[9];
    float sum = 0.f;
#pragma unroll
    for (int j = 0; j < 9; ++j) { e[j] = __expf(g[j] - mx); sum += e[j]; }
    const float inv = 1.f / sum;

    // Weighted sum of v neighborhood (zero padding contributes nothing).
    float o[CH_];
#pragma unroll
    for (int c = 0; c < CH_; ++c) o[c] = 0.f;
#pragma unroll
    for (int dk = 0; dk < 9; ++dk) {
        const int hh = h + dk / 3 - 1;
        const int ww = w + dk % 3 - 1;
        if (hh >= 0 && hh < H_ && ww >= 0 && ww < W_) {
            const int pk = hh * W_ + ww;
            const float sv = e[dk] * inv;
#pragma unroll
            for (int c = 0; c < CH_; ++c) o[c] += sv * vh[(size_t)c * HW_ + pk];
        }
    }

    // Store bf16 TRANSPOSED [b][hw][C]: 32 consecutive channels = 64
    // contiguous bytes per thread (16 packed dword stores).
    uint32_t* dst = (uint32_t*)(obT + ((size_t)b * HW_ + p) * C_ + head * CH_);
#pragma unroll
    for (int c = 0; c < CH_ / 2; ++c)
        dst[c] = (uint32_t)f2bf(o[2 * c]) | ((uint32_t)f2bf(o[2 * c + 1]) << 16);
}

// ---------------------------------------------------------------------------
// Kernel 4: output projection GEMM + bias.
// out[b, o, hw] = sum_c wo[o,c] * obT[b, hw, c] + bo[o]
// ---------------------------------------------------------------------------
__global__ __launch_bounds__(256) void out_gemm(const uint16_t* __restrict__ obT,
                                                const uint16_t* __restrict__ wob,
                                                const float* __restrict__ bo,
                                                float* __restrict__ out) {
    const int b = blockIdx.z;
    gemm_tile(wob,
              obT + (size_t)b * C_ * HW_,
              bo,
              out + (size_t)b * C_ * HW_,
              blockIdx.y, blockIdx.x);
}

// ---------------------------------------------------------------------------
// Host-side launcher.
// Inputs (setup_inputs order): q, k, v, wq, pe_dw, pe_pw, wo, bo  (all fp32)
// Output: fp32 (B, C, H, W)
// Workspace layout (all offsets 256B aligned):
//   [0      , 128KB)   wq bf16
//   [128KB  , 256KB)   wo bf16
//   [256KB  , +54MB)   xbT  bf16 activations, transposed: [src(3)][b][HW][C]
//   [...    , +108MB)  proj fp32:                          [src(3)][b][C][HW]
//   [...    , +18MB)   obT  bf16, transposed:              [b][HW][C]
// total ~ 180 MB
// ---------------------------------------------------------------------------
extern "C" void kernel_launch(void* const* d_in, const int* in_sizes, int n_in,
                              void* d_out, int out_size, void* d_ws, size_t ws_size,
                              hipStream_t stream) {
    const float* q     = (const float*)d_in[0];
    const float* k     = (const float*)d_in[1];
    const float* v     = (const float*)d_in[2];
    const float* wq    = (const float*)d_in[3];
    const float* pe_dw = (const float*)d_in[4];
    const float* pe_pw = (const float*)d_in[5];
    const float* wo    = (const float*)d_in[6];
    const float* bo    = (const float*)d_in[7];
    float* out = (float*)d_out;

    const size_t per_src = (size_t)B_ * C_ * HW_;       // elements per source

    char* ws = (char*)d_ws;
    uint16_t* wqb  = (uint16_t*)(ws);
    uint16_t* wob  = (uint16_t*)(ws + 131072);
    uint16_t* xbT  = (uint16_t*)(ws + 262144);
    float*    proj = (float*)   (ws + 262144 + 3 * per_src * sizeof(uint16_t));
    uint16_t* obT  = (uint16_t*)(ws + 262144 + 3 * per_src * sizeof(uint16_t)
                                           + 3 * per_src * sizeof(float));

    (void)in_sizes; (void)n_in; (void)out_size; (void)ws_size;

    // 1a. weights -> bf16
    cvt_weights<<<(C_ * C_ + 255) / 256, 256, 0, stream>>>(wq, wo, wqb, wob);

    // 1b. activations -> bf16, transposed to [hw][c] (LDS-tiled)
    cvt_acts_t<<<dim3(HW_ / 64, C_ / 64, 3 * B_), 256, 0, stream>>>(q, k, v, xbT);

    // 2. q/k/v projections (12 GEMM slices on the bf16 WMMA path)
    proj_gemm<<<dim3(HW_ / 128, C_ / 128, 3 * B_), 256, 0, stream>>>(xbT, wqb, proj);

    // 3. 3x3 local attention + PE + softmax, transposed bf16 output
    attn<<<(B_ * HEADS_ * HW_) / 256, 256, 0, stream>>>(proj, pe_dw, pe_pw, obT);

    // 4. output projection + bias
    out_gemm<<<dim3(HW_ / 128, C_ / 128, B_), 256, 0, stream>>>(obT, wob, bo, out);
}
